// vTransformer_2422361554952
// MI455X (gfx1250) — matrix operand order repaired and checked by reading, not compiled, not run on hardware
//
#include <hip/hip_runtime.h>
#include <hip/hip_bf16.h>

// ---------------------------------------------------------------------------
// Point Transformer layer for MI455X (gfx1250).
// B=4, IN_CH=128, N=2048, DIM=256, PH=64, HID=1024, K=16.
// Dominant compute: 1024x256 and 256x1024 GEMMs over 131072 columns
// -> bf16 WMMA 16x16x32, async-to-LDS double-buffered A tiles.
// ---------------------------------------------------------------------------

typedef __attribute__((ext_vector_type(2)))  __bf16 bf16x2;
typedef __attribute__((ext_vector_type(8)))  __bf16 bf16x8;
typedef __attribute__((ext_vector_type(16))) __bf16 bf16x16;
typedef __attribute__((ext_vector_type(8)))  float  f32x8;

// Pointer types for the async copy builtin (pointee must be 16B int vector).
typedef int v4i __attribute__((vector_size(16)));
typedef __attribute__((address_space(1))) v4i* gv4i_p;   // global
typedef __attribute__((address_space(3))) v4i* lv4i_p;   // LDS

#define DEV __device__ __forceinline__

static constexpr int BATCH  = 4;
static constexpr int IN_CH  = 128;
static constexpr int NPTS   = 2048;
static constexpr int DIMC   = 256;
static constexpr int PH     = 64;
static constexpr int HID    = 1024;
static constexpr int KNN    = 16;
static constexpr int NK     = NPTS * KNN;       // 32768
static constexpr float EPS  = 1e-5f;

// LDS tile row pitch (elements): 40*2B = 80B -> 16B-aligned rows; fragment
// ds_load_b128 reads cover all 64 banks in the minimal 2 passes.
static constexpr int LP = 40;

// ---- CDNA5 async global->LDS copy (ASYNCcnt path), guarded ----------------
#if defined(__has_builtin)
#  if __has_builtin(__builtin_amdgcn_global_load_async_to_lds_b128)
#    define USE_ASYNC_CP 1
#  endif
#endif
#ifndef USE_ASYNC_CP
#  define USE_ASYNC_CP 0
#endif

DEV void async_cp16(const __bf16* g, __bf16* l) {
#if USE_ASYNC_CP
  __builtin_amdgcn_global_load_async_to_lds_b128((gv4i_p)(g), (lv4i_p)(l), 0, 0);
#else
  *(bf16x8*)l = *(const bf16x8*)g;
#endif
}

template <int N> DEV void wait_async() {
#if USE_ASYNC_CP
#  if __has_builtin(__builtin_amdgcn_s_wait_asynccnt)
  __builtin_amdgcn_s_wait_asynccnt(N);
#  else
  if (N == 0) { asm volatile("s_wait_asynccnt 0x0" ::: "memory"); }
  else        { asm volatile("s_wait_asynccnt 0x2" ::: "memory"); }
#  endif
#endif
}

// ---- WMMA fragment loads (CDNA5 16x16x32 bf16 layouts, ISA 7.12.2) --------
// A (16x32, MxK): lane L: row = L%16, half = L/16.
//   element e -> K = 8*half + (e&7) + 16*(e>>3): two 16B chunks.
DEV bf16x16 load_a_frag(const __bf16* As, int row, int half) {
  const bf16x8 lo = *(const bf16x8*)(As + row * LP + 8 * half);
  const bf16x8 hi = *(const bf16x8*)(As + row * LP + 16 + 8 * half);
  return __builtin_shufflevector(lo, hi, 0,1,2,3,4,5,6,7,8,9,10,11,12,13,14,15);
}
// B (32x16, KxN) stored column-major in LDS (Bs[col][k]).
// lane L: col = L%16, half = L/16; element e -> K = 8*half + (e&7) + 16*(e>>3).
DEV bf16x16 load_b_frag(const __bf16* Bs, int col, int half) {
  const bf16x8 lo = *(const bf16x8*)(Bs + col * LP + 8 * half);
  const bf16x8 hi = *(const bf16x8*)(Bs + col * LP + 16 + 8 * half);
  return __builtin_shufflevector(lo, hi, 0,1,2,3,4,5,6,7,8,9,10,11,12,13,14,15);
}

// B staging helper: 2x2 micro-block transpose. The two loads per row are
// address-consecutive -> clang merges into b32 (bf16) / b64 (f32) loads;
// the two LDS stores are aligned 4B bf16x2.
template <typename TIN>
DEV void stage_b_2x2(const TIN* __restrict__ Bm, size_t src, size_t ncol,
                     __bf16* Bs, int n, int kp) {
  const TIN t00 = Bm[src],        t01 = Bm[src + 1];
  const TIN t10 = Bm[src + ncol], t11 = Bm[src + ncol + 1];
  bf16x2 c0, c1;
  c0.x = (__bf16)t00; c0.y = (__bf16)t10;   // column n:   K = kp, kp+1
  c1.x = (__bf16)t01; c1.y = (__bf16)t11;   // column n+1: K = kp, kp+1
  *(bf16x2*)(Bs + n * LP + kp) = c0;
  *(bf16x2*)(Bs + (n + 1) * LP + kp) = c1;
}

// ---------------------------------------------------------------------------
// Weight pre-pack: fp32 -> bf16 (row-major preserved).
// ---------------------------------------------------------------------------
__global__ __launch_bounds__(256) void cvt_f32_bf16(const float* __restrict__ s,
                                                    __bf16* __restrict__ d, int n) {
  const int i = blockIdx.x * 256 + threadIdx.x;
  if (i < n) d[i] = (__bf16)s[i];
}

// ---------------------------------------------------------------------------
// Generic WMMA GEMM:  C[M,Ncol] = act(A[M,Kd] * B[Kd,Ncol] + bias) per batch.
// A: bf16 row-major (pre-packed weights, shared across batch).
// Block: 256 thr = 8 waves (4M x 2N). Block tile 128x128, wave 32x64, K=32.
// A tiles double-buffered via async-to-LDS; B tiles VALU-transposed.
// EPI: 0 = bias; 1 = bias + BatchNorm + ReLU; 2 = bias + residual add.
// ---------------------------------------------------------------------------
template <typename TIN, typename TOUT, int EPI>
__global__ __launch_bounds__(256) void gemm_bf16_wmma(
    const __bf16* __restrict__ A, const TIN* __restrict__ Bm,
    TOUT* __restrict__ Cm, const float* __restrict__ bias,
    const float* __restrict__ g, const float* __restrict__ be,
    const float* __restrict__ mu, const float* __restrict__ va,
    const float* __restrict__ resid,
    int Kd, int Ncol, size_t bStride, size_t cStride) {
  __shared__ __bf16 As[2][128 * LP];
  __shared__ __bf16 Bs[128 * LP];

  const int tid  = threadIdx.x;
  const int bz   = blockIdx.z;
  const int row0 = blockIdx.y * 128;
  const int col0 = blockIdx.x * 128;
  const size_t bOff = (size_t)bz * bStride;
  const size_t cOff = (size_t)bz * cStride;

  const int wid = tid >> 5, lane = tid & 31;
  const int half = lane >> 4, l16 = lane & 15;
  const int wm = (wid & 3) * 32, wn = (wid >> 2) * 64;

  f32x8 acc[2][4] = {};

  // A tile (128x32 bf16) = 512 x 16B chunks; 2 async issues per thread.
  auto issueA = [&](int kb, int buf) {
#pragma unroll
    for (int i = 0; i < 2; ++i) {
      const int c = tid + i * 256;
      const int r = c >> 2, seg = (c & 3) << 3;
      async_cp16(A + (size_t)(row0 + r) * Kd + kb + seg, &As[buf][r * LP + seg]);
    }
  };

  issueA(0, 0);
  for (int kb = 0, ib = 0; kb < Kd; kb += 32, ++ib) {
    const int cur = ib & 1;
    const bool more = (kb + 32) < Kd;
    if (more) issueA(kb + 32, cur ^ 1);

    // Stage B tile (32 x 128) as 2x2 micro-blocks.
#pragma unroll
    for (int it = 0; it < 4; ++it) {
      const int i = tid + it * 256;            // 1024 (k-pair, n-pair) slots
      const int kp = (i >> 6) << 1, n = (i & 63) << 1;
      const size_t src = bOff + (size_t)(kb + kp) * Ncol + col0 + n;
      stage_b_2x2(Bm, src, (size_t)Ncol, Bs, n, kp);
    }
    if (more) wait_async<2>(); else wait_async<0>();
    __syncthreads();

    bf16x16 af[2], bfr[4];
#pragma unroll
    for (int mt = 0; mt < 2; ++mt) af[mt]  = load_a_frag(&As[cur][0], wm + 16 * mt + l16, half);
#pragma unroll
    for (int nt = 0; nt < 4; ++nt) bfr[nt] = load_b_frag(Bs, wn + 16 * nt + l16, half);
#pragma unroll
    for (int mt = 0; mt < 2; ++mt)
#pragma unroll
      for (int nt = 0; nt < 4; ++nt)
        acc[mt][nt] = __builtin_amdgcn_wmma_f32_16x16x32_bf16(
            false, af[mt], false, bfr[nt], (short)0, acc[mt][nt], false, false);
    __syncthreads();
  }

  // Epilogue. C layout: VGPR r -> M = r + 8*half, N = l16.
#pragma unroll
  for (int mt = 0; mt < 2; ++mt)
#pragma unroll
    for (int nt = 0; nt < 4; ++nt) {
      const int nG = col0 + wn + 16 * nt + l16;
#pragma unroll
      for (int r = 0; r < 8; ++r) {
        const int mG = row0 + wm + 16 * mt + r + 8 * half;
        float v = acc[mt][nt][r] + bias[mG];
        if (EPI == 1) {
          const float s = g[mG] * rsqrtf(va[mG] + EPS);
          v = fmaxf(v * s + (be[mG] - mu[mG] * s), 0.f);
        }
        if (EPI == 2) v += resid[cOff + (size_t)mG * Ncol + nG];
        Cm[cOff + (size_t)mG * Ncol + nG] = (TOUT)v;
      }
    }
}

// ---------------------------------------------------------------------------
// aw2 GEMM + fused softmax-over-K(=16) + attention aggregation.
// Block tile 128 x 128 -> 8 complete K-groups; logits never touch HBM.
// ---------------------------------------------------------------------------
__global__ __launch_bounds__(256) void gemm_attn_softmax_agg(
    const __bf16* __restrict__ A,       // aw2 packed (256 x 1024) bf16
    const __bf16* __restrict__ Bm,      // hidden (B,1024,NK) bf16
    const float* __restrict__ bias,     // ab2 (256)
    const float* __restrict__ value,    // (B,256,NPTS) f32
    const __bf16* __restrict__ pe,      // (B,256,NK) bf16
    const int* __restrict__ idx,        // (B,NPTS,16)
    __bf16* __restrict__ agg) {         // (B,256,NPTS) bf16
  __shared__ union SH {
    struct { __bf16 As[2][128 * LP]; __bf16 Bs[128 * LP]; } s;
    struct { float Cs[128 * 132]; int idxs[128]; } e;
  } sh;

  constexpr int Kd = HID, Ncol = NK;
  const int tid  = threadIdx.x;
  const int bz   = blockIdx.z;
  const int row0 = blockIdx.y * 128;
  const int col0 = blockIdx.x * 128;
  const size_t bOff = (size_t)bz * (size_t)HID * NK;

  const int wid = tid >> 5, lane = tid & 31;
  const int half = lane >> 4, l16 = lane & 15;
  const int wm = (wid & 3) * 32, wn = (wid >> 2) * 64;

  f32x8 acc[2][4] = {};

  auto issueA = [&](int kb, int buf) {
#pragma unroll
    for (int i = 0; i < 2; ++i) {
      const int c = tid + i * 256;
      const int r = c >> 2, seg = (c & 3) << 3;
      async_cp16(A + (size_t)(row0 + r) * Kd + kb + seg, &sh.s.As[buf][r * LP + seg]);
    }
  };

  issueA(0, 0);
  for (int kb = 0, ib = 0; kb < Kd; kb += 32, ++ib) {
    const int cur = ib & 1;
    const bool more = (kb + 32) < Kd;
    if (more) issueA(kb + 32, cur ^ 1);
#pragma unroll
    for (int it = 0; it < 4; ++it) {
      const int i = tid + it * 256;
      const int kp = (i >> 6) << 1, n = (i & 63) << 1;
      const size_t src = bOff + (size_t)(kb + kp) * Ncol + col0 + n;
      stage_b_2x2(Bm, src, (size_t)Ncol, sh.s.Bs, n, kp);
    }
    if (more) wait_async<2>(); else wait_async<0>();
    __syncthreads();

    bf16x16 af[2], bfr[4];
#pragma unroll
    for (int mt = 0; mt < 2; ++mt) af[mt]  = load_a_frag(&sh.s.As[cur][0], wm + 16 * mt + l16, half);
#pragma unroll
    for (int nt = 0; nt < 4; ++nt) bfr[nt] = load_b_frag(sh.s.Bs, wn + 16 * nt + l16, half);
#pragma unroll
    for (int mt = 0; mt < 2; ++mt)
#pragma unroll
      for (int nt = 0; nt < 4; ++nt)
        acc[mt][nt] = __builtin_amdgcn_wmma_f32_16x16x32_bf16(
            false, af[mt], false, bfr[nt], (short)0, acc[mt][nt], false, false);
    __syncthreads();
  }

  // Spill logits (+bias) to LDS (overlaps stage buffers; barrier above).
#pragma unroll
  for (int mt = 0; mt < 2; ++mt)
#pragma unroll
    for (int nt = 0; nt < 4; ++nt) {
      const int nl = wn + 16 * nt + l16;
#pragma unroll
      for (int r = 0; r < 8; ++r) {
        const int ml = wm + 16 * mt + r + 8 * half;
        sh.e.Cs[ml * 132 + nl] = acc[mt][nt][r] + bias[row0 + ml];
      }
    }
  if (tid < 128) sh.e.idxs[tid] = idx[(size_t)bz * NK + col0 + tid];
  __syncthreads();

  // 1024 (row, point) pairs: softmax over 16 + gather-weighted sum.
  for (int q = tid; q < 1024; q += 256) {
    const int ml = q >> 3, p = q & 7;
    const float* row = sh.e.Cs + ml * 132 + p * 16;
    float mx = row[0];
#pragma unroll
    for (int k = 1; k < 16; ++k) mx = fmaxf(mx, row[k]);
    float ex[16], s = 0.f;
#pragma unroll
    for (int k = 0; k < 16; ++k) { ex[k] = __expf(row[k] - mx); s += ex[k]; }
    const float inv = 1.f / s;
    const int mg = row0 + ml;
    const float*  vrow = value + ((size_t)bz * DIMC + mg) * NPTS;
    const __bf16* prow = pe + ((size_t)bz * DIMC + mg) * NK + col0 + p * 16;
    float a = 0.f;
#pragma unroll
    for (int k = 0; k < 16; ++k)
      a += ex[k] * inv * (vrow[sh.e.idxs[p * 16 + k]] + (float)prow[k]);
    agg[((size_t)bz * DIMC + mg) * NPTS + (col0 >> 4) + p] = (__bf16)a;
  }
}

// ---------------------------------------------------------------------------
// KNN: per query point, top-16 smallest squared distances (self included).
// ---------------------------------------------------------------------------
__global__ __launch_bounds__(256) void knn_kernel(const float* __restrict__ pos,
                                                  int* __restrict__ idx) {
  __shared__ float sx[NPTS], sy[NPTS], sz[NPTS];
  const int bz = blockIdx.y, tid = threadIdx.x;
  const float* pb = pos + (size_t)bz * 3 * NPTS;
  for (int i = tid; i < NPTS; i += 256) {
    sx[i] = pb[i]; sy[i] = pb[NPTS + i]; sz[i] = pb[2 * NPTS + i];
  }
  __syncthreads();
  const int n = blockIdx.x * 256 + tid;
  const float qx = sx[n], qy = sy[n], qz = sz[n];
  float bd[16]; int bi[16];
#pragma unroll
  for (int s = 0; s < 16; ++s) { bd[s] = 3.4e38f; bi[s] = 0; }
  for (int j = 0; j < NPTS; ++j) {
    const float dx = qx - sx[j], dy = qy - sy[j], dz = qz - sz[j];
    const float d = dx * dx + dy * dy + dz * dz;
    if (d < bd[15]) {
#pragma unroll
      for (int s = 15; s >= 0; --s) {       // ascending insertion, stable ties
        if (s == 0 || !(d < bd[s - 1])) { bd[s] = d; bi[s] = j; break; }
        bd[s] = bd[s - 1]; bi[s] = bi[s - 1];
      }
    }
  }
  int* o = idx + ((size_t)bz * NPTS + n) * KNN;
#pragma unroll
  for (int s = 0; s < 16; ++s) o[s] = bi[s];
}

// ---------------------------------------------------------------------------
// pe hidden layer: pos_rel (3) -> PH=64 with BN+ReLU, bf16 out (B,64,NK).
// ---------------------------------------------------------------------------
__global__ __launch_bounds__(256) void pe_hidden_kernel(
    const float* __restrict__ pos, const int* __restrict__ idx,
    const float* __restrict__ pw1, const float* __restrict__ pb1,
    const float* __restrict__ pg, const float* __restrict__ pbeta,
    const float* __restrict__ pmean, const float* __restrict__ pvar,
    __bf16* __restrict__ ph) {
  const int bz = blockIdx.y;
  const int col = blockIdx.x * 256 + threadIdx.x;   // n*16+k
  const int n = col >> 4;
  const int jn = idx[(size_t)bz * NK + col];
  const float* pb = pos + (size_t)bz * 3 * NPTS;
  const float px = pb[n] - pb[jn];
  const float py = pb[NPTS + n] - pb[NPTS + jn];
  const float pz = pb[2 * NPTS + n] - pb[2 * NPTS + jn];
  __bf16* out = ph + (size_t)bz * PH * NK + col;
#pragma unroll 8
  for (int j = 0; j < PH; ++j) {
    const float d = pw1[j * 3] * px + pw1[j * 3 + 1] * py + pw1[j * 3 + 2] * pz + pb1[j];
    const float s = pg[j] * rsqrtf(pvar[j] + EPS);
    out[(size_t)j * NK] = (__bf16)fmaxf(d * s + (pbeta[j] - pmean[j] * s), 0.f);
  }
}

// ---------------------------------------------------------------------------
// attn input: query[c,n] - key[c,idx[n,k]] + pe[c,n,k]  -> bf16 (B,256,NK).
// ---------------------------------------------------------------------------
__global__ __launch_bounds__(256) void attn_in_kernel(
    const float* __restrict__ query, const float* __restrict__ keyc,
    const int* __restrict__ idx, const __bf16* __restrict__ pe,
    __bf16* __restrict__ attn_in) {
  const int bz = blockIdx.z, c = blockIdx.y;
  const int col = blockIdx.x * 256 + threadIdx.x;
  const int n = col >> 4;
  const int jn = idx[(size_t)bz * NK + col];
  const size_t rc = (size_t)bz * DIMC + c;
  const float v = query[rc * NPTS + n] - keyc[rc * NPTS + jn] + (float)pe[rc * NK + col];
  attn_in[rc * NK + col] = (__bf16)v;
}

// ---------------------------------------------------------------------------
extern "C" void kernel_launch(void* const* d_in, const int* in_sizes, int n_in,
                              void* d_out, int out_size, void* d_ws, size_t ws_size,
                              hipStream_t stream) {
  (void)in_sizes; (void)n_in; (void)out_size; (void)ws_size;
  const float* x     = (const float*)d_in[0];
  const float* pos   = (const float*)d_in[1];
  const float* ls_w  = (const float*)d_in[2];
  const float* ls_b  = (const float*)d_in[3];
  const float* kw    = (const float*)d_in[4];
  const float* kb    = (const float*)d_in[5];
  const float* qw    = (const float*)d_in[6];
  const float* qb    = (const float*)d_in[7];
  const float* vw    = (const float*)d_in[8];
  const float* vb    = (const float*)d_in[9];
  const float* pw1   = (const float*)d_in[10];
  const float* pb1   = (const float*)d_in[11];
  const float* pg    = (const float*)d_in[12];
  const float* pbeta = (const float*)d_in[13];
  const float* pmean = (const float*)d_in[14];
  const float* pvar  = (const float*)d_in[15];
  const float* pw2   = (const float*)d_in[16];
  const float* pb2   = (const float*)d_in[17];
  const float* aw1   = (const float*)d_in[18];
  const float* ab1   = (const float*)d_in[19];
  const float* ag    = (const float*)d_in[20];
  const float* abeta = (const float*)d_in[21];
  const float* amean = (const float*)d_in[22];
  const float* avar  = (const float*)d_in[23];
  const float* aw2   = (const float*)d_in[24];
  const float* ab2   = (const float*)d_in[25];
  const float* lw    = (const float*)d_in[26];
  const float* lb    = (const float*)d_in[27];

  char* ws = (char*)d_ws;
  size_t off = 0;
  auto carve = [&](size_t bytes) -> void* {
    void* p = ws + off;
    off = (off + bytes + 255) & ~(size_t)255;
    return p;
  };
  // Packed bf16 weights.
  __bf16* p_ls  = (__bf16*)carve((size_t)DIMC * IN_CH * 2);
  __bf16* p_kw  = (__bf16*)carve((size_t)DIMC * DIMC * 2);
  __bf16* p_qw  = (__bf16*)carve((size_t)DIMC * DIMC * 2);
  __bf16* p_vw  = (__bf16*)carve((size_t)DIMC * DIMC * 2);
  __bf16* p_pw2 = (__bf16*)carve((size_t)DIMC * PH * 2);
  __bf16* p_aw1 = (__bf16*)carve((size_t)HID * DIMC * 2);
  __bf16* p_aw2 = (__bf16*)carve((size_t)DIMC * HID * 2);
  __bf16* p_lw  = (__bf16*)carve((size_t)IN_CH * DIMC * 2);
  // Activations / intermediates.
  __bf16* h_bf    = (__bf16*)carve((size_t)BATCH * DIMC * NPTS * 2);
  float*  key_f   = (float*)carve((size_t)BATCH * DIMC * NPTS * 4);
  float*  query_f = (float*)carve((size_t)BATCH * DIMC * NPTS * 4);
  float*  value_f = (float*)carve((size_t)BATCH * DIMC * NPTS * 4);
  int*    idx_i   = (int*)carve((size_t)BATCH * NK * 4);
  __bf16* pe_hid  = (__bf16*)carve((size_t)BATCH * PH * NK * 2);
  __bf16* pe_bf   = (__bf16*)carve((size_t)BATCH * DIMC * NK * 2);
  __bf16* attn_in = (__bf16*)carve((size_t)BATCH * DIMC * NK * 2);
  __bf16* hidden  = (__bf16*)carve((size_t)BATCH * HID * NK * 2);
  __bf16* agg_bf  = (__bf16*)carve((size_t)BATCH * DIMC * NPTS * 2);

  // 0) Pre-pack weights to bf16 (enables raw async A-tile copies).
  auto pack = [&](const float* s, __bf16* d, int n) {
    cvt_f32_bf16<<<dim3((n + 255) / 256), 256, 0, stream>>>(s, d, n);
  };
  pack(ls_w, p_ls, DIMC * IN_CH);
  pack(kw, p_kw, DIMC * DIMC);
  pack(qw, p_qw, DIMC * DIMC);
  pack(vw, p_vw, DIMC * DIMC);
  pack(pw2, p_pw2, DIMC * PH);
  pack(aw1, p_aw1, HID * DIMC);
  pack(aw2, p_aw2, DIMC * HID);
  pack(lw, p_lw, IN_CH * DIMC);

  // 1) KNN indices.
  knn_kernel<<<dim3(NPTS / 256, BATCH), 256, 0, stream>>>(pos, idx_i);

  // 2) h = ls_w @ x + ls_b  (256x128 @ 128x2048) -> bf16.
  gemm_bf16_wmma<float, __bf16, 0><<<dim3(NPTS / 128, DIMC / 128, BATCH), 256, 0, stream>>>(
      p_ls, x, h_bf, ls_b, nullptr, nullptr, nullptr, nullptr, nullptr,
      IN_CH, NPTS, (size_t)IN_CH * NPTS, (size_t)DIMC * NPTS);

  // 3) key / query / value = {k,q,v}w @ h  (256x256 @ 256x2048) -> f32.
  gemm_bf16_wmma<__bf16, float, 0><<<dim3(NPTS / 128, DIMC / 128, BATCH), 256, 0, stream>>>(
      p_kw, h_bf, key_f, kb, nullptr, nullptr, nullptr, nullptr, nullptr,
      DIMC, NPTS, (size_t)DIMC * NPTS, (size_t)DIMC * NPTS);
  gemm_bf16_wmma<__bf16, float, 0><<<dim3(NPTS / 128, DIMC / 128, BATCH), 256, 0, stream>>>(
      p_qw, h_bf, query_f, qb, nullptr, nullptr, nullptr, nullptr, nullptr,
      DIMC, NPTS, (size_t)DIMC * NPTS, (size_t)DIMC * NPTS);
  gemm_bf16_wmma<__bf16, float, 0><<<dim3(NPTS / 128, DIMC / 128, BATCH), 256, 0, stream>>>(
      p_vw, h_bf, value_f, vb, nullptr, nullptr, nullptr, nullptr, nullptr,
      DIMC, NPTS, (size_t)DIMC * NPTS, (size_t)DIMC * NPTS);

  // 4) pe hidden: pos_rel -> 64ch MLP + BN + ReLU (bf16).
  pe_hidden_kernel<<<dim3(NK / 256, BATCH), 256, 0, stream>>>(
      pos, idx_i, pw1, pb1, pg, pbeta, pmean, pvar, pe_hid);

  // 5) pe = pw2 @ pe_hid + pb2  (256x64 @ 64x32768) -> bf16.
  gemm_bf16_wmma<__bf16, __bf16, 0><<<dim3(NK / 128, DIMC / 128, BATCH), 256, 0, stream>>>(
      p_pw2, pe_hid, pe_bf, pb2, nullptr, nullptr, nullptr, nullptr, nullptr,
      PH, NK, (size_t)PH * NK, (size_t)DIMC * NK);

  // 6) attn input = query - gathered key + pe (bf16).
  attn_in_kernel<<<dim3(NK / 256, DIMC, BATCH), 256, 0, stream>>>(
      query_f, key_f, idx_i, pe_bf, attn_in);

  // 7) hidden = relu(bn(aw1 @ attn_in + ab1))  (1024x256 @ 256x32768) -> bf16.
  gemm_bf16_wmma<__bf16, __bf16, 1><<<dim3(NK / 128, HID / 128, BATCH), 256, 0, stream>>>(
      p_aw1, attn_in, hidden, ab1, ag, abeta, amean, avar, nullptr,
      DIMC, NK, (size_t)DIMC * NK, (size_t)HID * NK);

  // 8) aw2 GEMM + softmax(K=16) + weighted aggregation -> agg (bf16).
  gemm_attn_softmax_agg<<<dim3(NK / 128, DIMC / 128, BATCH), 256, 0, stream>>>(
      p_aw2, hidden, ab2, value_f, pe_bf, idx_i, agg_bf);

  // 9) y = lw @ agg + lb + x  (128x256 @ 256x2048) -> f32 out with residual.
  gemm_bf16_wmma<__bf16, float, 2><<<dim3(NPTS / 128, IN_CH / 128, BATCH), 256, 0, stream>>>(
      p_lw, agg_bf, (float*)d_out, lb, nullptr, nullptr, nullptr, nullptr, x,
      DIMC, NPTS, (size_t)DIMC * NPTS, (size_t)IN_CH * NPTS);
}
